// BackgroundNoiseLayer_34170759807366
// MI455X (gfx1250) — compile-verified, matrix operand order checked by reading, not run on hardware
//
#include <hip/hip_runtime.h>

typedef __bf16 bf16_t;
typedef bf16_t v16bf __attribute__((ext_vector_type(16)));
typedef bf16_t v8bf  __attribute__((ext_vector_type(8)));
typedef float  v8f   __attribute__((ext_vector_type(8)));
typedef float  v4f   __attribute__((ext_vector_type(4)));

#define K_DIM     100
#define K_PAD     128
#define NNEUR     50000
#define R_BASIS   5
#define C_COLS    (NNEUR * R_BASIS)   /* 250000 output columns (n*5+r) */
#define P_SPIKE   0.25f

// ---------------- kernel 1: zero the f32 W accumulation buffer ----------------
__global__ void zero_ws(float4* __restrict__ p, long long n4) {
  long long i = (long long)blockIdx.x * blockDim.x + threadIdx.x;
  long long stride = (long long)gridDim.x * blockDim.x;
  float4 z = {0.f, 0.f, 0.f, 0.f};
  for (; i < n4; i += stride) p[i] = z;
}

// ---------------- kernel 2: bf16 spike matrix (t_pad x 128) ----------------
__global__ void make_spikes(const float* __restrict__ u, bf16_t* __restrict__ spikes,
                            int BT, int TP) {
  int idx = blockIdx.x * blockDim.x + threadIdx.x;
  int total = TP * K_PAD;
  if (idx >= total) return;
  int t = idx >> 7;          // / K_PAD
  int k = idx & (K_PAD - 1); // % K_PAD
  float v = 0.f;
  if (t < BT && k < K_DIM) v = (u[t * K_DIM + k] < P_SPIKE) ? 1.0f : 0.0f;
  spikes[idx] = (bf16_t)v;
}

// ---------------- kernel 3: sparse scatter-add into Wt[c][k], c = n*5+r ------
__global__ void scatter_w(const int* __restrict__ rows, const int* __restrict__ cols,
                          const float* __restrict__ w, const float* __restrict__ wf,
                          float* __restrict__ Wt, int nnz) {
  int e = blockIdx.x * blockDim.x + threadIdx.x;
  if (e >= nnz) return;
  int n = rows[e];
  int k = cols[e];
  float wv = w[e];
  long long base = (long long)n * R_BASIS * K_PAD + k;
#pragma unroll
  for (int r = 0; r < R_BASIS; ++r) {
    atomicAdd(&Wt[base + (long long)r * K_PAD], wv * wf[e * R_BASIS + r]);
  }
}

// ---------------- kernel 4: WMMA GEMM: out(t,c) = sum_k spikes(t,k)*Wt(c,k) ---
__global__ void __launch_bounds__(128)
spmm_wmma(const bf16_t* __restrict__ spikes, const float* __restrict__ Wt,
          float* __restrict__ out, int BT, int nTTiles) {
  const int nStrips = C_COLS / 16;                       // 15625
  int wave = blockIdx.x * (blockDim.x >> 5) + (threadIdx.x >> 5);
  if (wave >= nStrips) return;                           // whole wave exits -> EXEC full
  int c0   = wave * 16;
  int lane = threadIdx.x & 31;
  int col  = lane & 15;                                  // B column / A row within tile
  int half = lane >> 4;                                  // K sub-offset selector

  // ---- preload the 4 B fragments (K=128 as 4 steps of 32), held in VGPRs ----
  // B layout (32x16, 16-bit): lane holds column `col`; elems 0..7 = K = k0..k0+7,
  // elems 8..15 = K = k0+16..k0+23, with k0 = kb*32 + half*8.
  // Wt is c-major, so a lane's column chunk is a contiguous 128B cacheline.
  v16bf bfrag[4];
  const float* bcol = Wt + (long long)(c0 + col) * K_PAD;
#pragma unroll
  for (int kb = 0; kb < 4; ++kb) {
    int k0 = kb * 32 + half * 8;
    v4f x0 = *(const v4f*)(bcol + k0);
    v4f x1 = *(const v4f*)(bcol + k0 + 4);
    v4f x2 = *(const v4f*)(bcol + k0 + 16);
    v4f x3 = *(const v4f*)(bcol + k0 + 20);
#pragma unroll
    for (int i = 0; i < 4; ++i) {
      bfrag[kb][i]      = (bf16_t)x0[i];
      bfrag[kb][4 + i]  = (bf16_t)x1[i];
      bfrag[kb][8 + i]  = (bf16_t)x2[i];
      bfrag[kb][12 + i] = (bf16_t)x3[i];
    }
  }

  // A layout (16x32, 16-bit): lane holds row M = lane&15; elems 0..7 = K k0..k0+7,
  // elems 8..15 = K k0+16..k0+23, k0 = kb*32 + half*8.
  const bf16_t* abase = spikes + (long long)col * K_PAD;
  const long long tileStride = 16LL * K_PAD;

  // prime the A double buffer with tile 0
  v8bf alo[4], ahi[4];
#pragma unroll
  for (int kb = 0; kb < 4; ++kb) {
    int k0 = kb * 32 + half * 8;
    alo[kb] = *(const v8bf*)(abase + k0);
    ahi[kb] = *(const v8bf*)(abase + k0 + 16);
  }

  const int nFull = nTTiles - 1;  // all tiles except the last are fully in-bounds
  for (int tt = 0; tt < nFull; ++tt) {
    // prefetch next A tile while this tile computes/stores
    const bf16_t* anext = abase + (long long)(tt + 1) * tileStride;
    v8bf nlo[4], nhi[4];
#pragma unroll
    for (int kb = 0; kb < 4; ++kb) {
      int k0 = kb * 32 + half * 8;
      nlo[kb] = *(const v8bf*)(anext + k0);
      nhi[kb] = *(const v8bf*)(anext + k0 + 16);
    }

    v8f acc = {};
#pragma unroll
    for (int kb = 0; kb < 4; ++kb) {
      v16bf afrag;
#pragma unroll
      for (int i = 0; i < 8; ++i) { afrag[i] = alo[kb][i]; afrag[8 + i] = ahi[kb][i]; }
      acc = __builtin_amdgcn_wmma_f32_16x16x32_bf16(
          false, afrag, false, bfrag[kb], (short)0, acc, false, false);
    }

    // D layout: VGPR v -> row M = v + half*8, column N = lane&15.
    // Unconditional, non-temporal streaming stores (output never re-read;
    // keeps Wt/spikes resident in the 192MB L2).
    float* obase = out + (long long)(tt * 16 + half * 8) * C_COLS + c0 + col;
#pragma unroll
    for (int v = 0; v < 8; ++v) {
      __builtin_nontemporal_store(acc[v], obase + (long long)v * C_COLS);
    }

#pragma unroll
    for (int kb = 0; kb < 4; ++kb) { alo[kb] = nlo[kb]; ahi[kb] = nhi[kb]; }
  }

  // ---- epilogue: last (possibly partial) t-tile with predicated stores ----
  {
    int t0 = nFull * 16;
    v8f acc = {};
#pragma unroll
    for (int kb = 0; kb < 4; ++kb) {
      v16bf afrag;
#pragma unroll
      for (int i = 0; i < 8; ++i) { afrag[i] = alo[kb][i]; afrag[8 + i] = ahi[kb][i]; }
      acc = __builtin_amdgcn_wmma_f32_16x16x32_bf16(
          false, afrag, false, bfrag[kb], (short)0, acc, false, false);
    }
#pragma unroll
    for (int v = 0; v < 8; ++v) {
      int t = t0 + v + half * 8;
      if (t < BT) {
        __builtin_nontemporal_store(acc[v], out + (long long)t * C_COLS + c0 + col);
      }
    }
  }
}

extern "C" void kernel_launch(void* const* d_in, const int* in_sizes, int n_in,
                              void* d_out, int out_size, void* d_ws, size_t ws_size,
                              hipStream_t stream) {
  const float* u    = (const float*)d_in[0];
  const int*   rows = (const int*)d_in[1];
  const int*   cols = (const int*)d_in[2];
  const float* w    = (const float*)d_in[3];
  const float* wf   = (const float*)d_in[4];
  float* out = (float*)d_out;

  int BT  = in_sizes[0] / K_DIM;            // 600
  int nnz = in_sizes[1];                    // 200000
  int TP  = ((BT + 15) / 16) * 16;          // 608
  int nTTiles = TP / 16;                    // 38

  // workspace: [ Wt : C_COLS*K_PAD f32 (128 MB) ][ spikes : TP*K_PAD bf16 ]
  float*  Wt = (float*)d_ws;
  size_t wt_bytes = (size_t)C_COLS * K_PAD * sizeof(float);
  bf16_t* spikes = (bf16_t*)((char*)d_ws + wt_bytes);

  long long n4 = (long long)(wt_bytes / 16);
  zero_ws<<<2048, 256, 0, stream>>>((float4*)d_ws, n4);
  make_spikes<<<(TP * K_PAD + 255) / 256, 256, 0, stream>>>(u, spikes, BT, TP);
  scatter_w<<<(nnz + 255) / 256, 256, 0, stream>>>(rows, cols, w, wf, Wt, nnz);

  const int nStrips = C_COLS / 16;          // 15625 waves
  const int wavesPerBlock = 4;              // 128 threads = 4 wave32
  int blocks = (nStrips + wavesPerBlock - 1) / wavesPerBlock;
  spmm_wmma<<<blocks, 128, 0, stream>>>(spikes, Wt, out, BT, nTTiles);
}
